// SCRO_41231686042264
// MI455X (gfx1250) — compile-verified, matrix operand order checked by reading, not run on hardware
//
#include <hip/hip_runtime.h>
#include <hip/hip_bf16.h>

// ---------------------------------------------------------------------------
// MI455X (gfx1250) implementation.
// Dominant cost: hidden(BS,768) x gate_wh^T(768,768)  ~38.7 GFLOP, plus
// hs(BS,768) x lin_w^T(768,98) ~4.9 GFLOP.  Both routed through
// v_wmma_f32_16x16x32_bf16 (bf16 in, f32 accum).  Memory floor ~115 MB
// (~5-10us @23.3TB/s); bf16 WMMA keeps compute in the same ballpark.
// ---------------------------------------------------------------------------

typedef __attribute__((ext_vector_type(16))) __bf16 v16bf;
typedef __attribute__((ext_vector_type(8)))  __bf16 v8bf;
typedef __attribute__((ext_vector_type(8)))  float  v8f;

union FragA { v16bf v; v8bf h[2]; };

#define Bb   32
#define Ss   1024
#define Hh   768
#define P2   98
#define LDA  776   // 768 + 8 bf16 pad: row stride 1552B, 1552%256==16 -> no bank conflicts

__device__ __forceinline__ unsigned short f2bf(float f) {
  unsigned u = __builtin_bit_cast(unsigned, f);
  u += 0x7FFFu + ((u >> 16) & 1u);                  // round-to-nearest-even
  return (unsigned short)(u >> 16);
}
__device__ __forceinline__ float bf2f(unsigned short s) {
  unsigned u = ((unsigned)s) << 16;
  return __builtin_bit_cast(float, u);
}
__device__ __forceinline__ float sigmoidf(float x) {
  return 1.0f / (1.0f + __expf(-x));
}

// ---------------------------------------------------------------------------
// Phase 1: per-batch masked softmax attention pooling -> s_x (B,H), cue_len
// One 256-thread workgroup per batch. Bandwidth-bound.
// ---------------------------------------------------------------------------
__global__ __launch_bounds__(256) void k_attn(const float* __restrict__ hs,
                                              const int* __restrict__ mask,
                                              const float* __restrict__ attn_w,
                                              const float* __restrict__ attn_b,
                                              float* __restrict__ sx,
                                              int* __restrict__ cue) {
  __shared__ float sw[Hh];
  __shared__ float sc[Ss];
  __shared__ float red[256];
  const int b = blockIdx.x, t = threadIdx.x;

  for (int h = t; h < Hh; h += 256) sw[h] = attn_w[h];

  int cnt = 0;
  for (int s = t; s < Ss; s += 256) cnt += mask[b * Ss + s];
  red[t] = (float)cnt;
  __syncthreads();
  for (int o = 128; o > 0; o >>= 1) { if (t < o) red[t] += red[t + o]; __syncthreads(); }
  int cueLen = (int)red[0];
  cueLen = cueLen < 1 ? 1 : (cueLen > Ss ? Ss : cueLen);

  const float ab = attn_b[0];
  for (int s = t; s < Ss; s += 256) {
    const float* row = hs + ((size_t)b * Ss + s) * Hh;
    float acc = 0.0f;
    for (int h = 0; h < Hh; h += 4) {
      acc += row[h] * sw[h] + row[h + 1] * sw[h + 1] +
             row[h + 2] * sw[h + 2] + row[h + 3] * sw[h + 3];
    }
    sc[s] = (s < cueLen) ? (acc + ab) : -3.0e38f;
  }
  __syncthreads();

  float lm = -3.0e38f;
  for (int s = t; s < Ss; s += 256) lm = fmaxf(lm, sc[s]);
  red[t] = lm;
  __syncthreads();
  for (int o = 128; o > 0; o >>= 1) { if (t < o) red[t] = fmaxf(red[t], red[t + o]); __syncthreads(); }
  const float mx = red[0];
  __syncthreads();

  float ls = 0.0f;
  for (int s = t; s < Ss; s += 256) {
    float e = (s < cueLen) ? __expf(sc[s] - mx) : 0.0f;
    sc[s] = e;
    ls += e;
  }
  red[t] = ls;
  __syncthreads();
  for (int o = 128; o > 0; o >>= 1) { if (t < o) red[t] += red[t + o]; __syncthreads(); }
  const float inv = 1.0f / red[0];
  __syncthreads();
  for (int s = t; s < Ss; s += 256) sc[s] *= inv;
  __syncthreads();

  // s_x[h] = sum_s attn[s]*hidden[b,s,h]  (coalesced over h across threads)
  for (int h = t; h < Hh; h += 256) {
    float acc = 0.0f;
    for (int s = 0; s < cueLen; ++s) acc += sc[s] * hs[((size_t)b * Ss + s) * Hh + h];
    sx[b * Hh + h] = acc;
  }
  if (t == 0) cue[b] = cueLen;
}

// ---------------------------------------------------------------------------
// Phase 1b: g_s[b,o] = sum_h s_x[b,h]*gate_w[o,h] + gate_b[o]   (tiny: ~38 MFLOP)
// ---------------------------------------------------------------------------
__global__ __launch_bounds__(256) void k_gs(const float* __restrict__ sx,
                                            const float* __restrict__ gw,
                                            const float* __restrict__ gb,
                                            float* __restrict__ gs) {
  __shared__ float s[Hh];
  const int b = blockIdx.x, t = threadIdx.x;
  const int o = blockIdx.y * 256 + t;
  for (int h = t; h < Hh; h += 256) s[h] = sx[b * Hh + h];
  __syncthreads();
  const float* wr = gw + (size_t)o * (2 * Hh);    // gate_ws = gate_w[:, :H]
  float acc = gb[o];
  for (int h = 0; h < Hh; ++h) acc += s[h] * wr[h];
  gs[b * Hh + o] = acc;
}

// ---------------------------------------------------------------------------
// Weight packing: emit bf16 B-fragments in exact per-lane WMMA register order
// (ISA 7.12.2, 16-bit operands).  B[k][n] lives where A[n][k] would:
//   lane<16 : n=lane,    k in {0..7, 16..23}  (j 0..7 -> k=j ; j 8..15 -> k=j+8)
//   lane>=16: n=lane-16, k offsets +8
// Main kernel then loads one contiguous 32B v16bf per lane -> perfect coalescing.
// ---------------------------------------------------------------------------
__global__ __launch_bounds__(256) void k_pack_gate(const float* __restrict__ gw,
                                                   unsigned short* __restrict__ pg) {
  const int g = blockIdx.x * 256 + threadIdx.x;
  if (g >= 24 * 48 * 512) return;
  const int tile = g >> 9, r = g & 511;
  const int L = r >> 4, j = r & 15;
  const int kt = tile / 48, nt = tile % 48;
  const int k = kt * 32 + ((j < 8) ? j : j + 8) + ((L >= 16) ? 8 : 0);
  const int n = nt * 16 + (L & 15);
  pg[g] = f2bf(gw[(size_t)n * (2 * Hh) + Hh + k]);   // gate_wh = gate_w[:, H:]
}

__global__ __launch_bounds__(256) void k_pack_lin(const float* __restrict__ lw,
                                                  unsigned short* __restrict__ pl) {
  const int g = blockIdx.x * 256 + threadIdx.x;
  if (g >= 24 * 7 * 512) return;
  const int tile = g >> 9, r = g & 511;
  const int L = r >> 4, j = r & 15;
  const int kt = tile / 7, nt = tile % 7;
  const int k = kt * 32 + ((j < 8) ? j : j + 8) + ((L >= 16) ? 8 : 0);
  const int n = nt * 16 + (L & 15);
  pl[g] = f2bf((n < P2) ? lw[(size_t)n * Hh + k] : 0.0f);
}

// ---------------------------------------------------------------------------
// Phase 2: main fused kernel. One WG (8 wave32) per 16-token tile.
//   gate GEMM (16x768x768) -> sigmoid gate -> hs tile (bf16, LDS)
//   lin  GEMM (16x112x768) -> sigmoid+clip -> out
// ---------------------------------------------------------------------------
__global__ __launch_bounds__(256) void k_main(const float* __restrict__ hs,
                                              const int* __restrict__ cue,
                                              const float* __restrict__ sx,
                                              const float* __restrict__ gs,
                                              const unsigned short* __restrict__ pGate,
                                              const unsigned short* __restrict__ pLin,
                                              const float* __restrict__ lin_b,
                                              float* __restrict__ out) {
  __shared__ unsigned short shA[16 * LDA];   // hidden tile, bf16
  __shared__ unsigned short shHS[16 * LDA];  // hs tile, bf16
  __shared__ float shSX[Hh];
  __shared__ float shGS[Hh];

  const int b  = blockIdx.x >> 6;            // 64 tiles per batch (S/16)
  const int s0 = (blockIdx.x & 63) << 4;
  const int t  = threadIdx.x;
  const int wave = t >> 5, lane = t & 31;
  const int cueLen = cue[b];

  // ---- stage hidden tile (16x768 f32 -> bf16 in LDS) + per-batch vectors ----
  const float* tileP = hs + (((size_t)b * Ss) + s0) * Hh;
  for (int q = t; q < 3072; q += 256) {        // 3072 float4 = 16*768 floats
    const int m  = q / 192;
    const int q4 = q % 192;
    const float4 v = ((const float4*)(tileP + (size_t)m * Hh))[q4];
    unsigned short* d = &shA[m * LDA + q4 * 4];
    *(unsigned*)(d)     = (unsigned)f2bf(v.x) | ((unsigned)f2bf(v.y) << 16);
    *(unsigned*)(d + 2) = (unsigned)f2bf(v.z) | ((unsigned)f2bf(v.w) << 16);
  }
  for (int h = t; h < Hh; h += 256) {
    shSX[h] = sx[b * Hh + h];
    shGS[h] = gs[b * Hh + h];
  }
  __syncthreads();

  const int mRow = lane & 15;                  // A-fragment row for this lane
  const int kSel = (lane >= 16) ? 8 : 0;       // A-fragment k-chunk select
  const int mHi  = (lane >= 16) ? 8 : 0;       // C/D-fragment M offset

  // ---- gate GEMM: each wave owns 6 N-tiles (48 total), K = 24 steps ----
  for (int tt = 0; tt < 6; ++tt) {
    const int nt = wave * 6 + tt;
    v8f acc = {};
    const v16bf* bp = (const v16bf*)pGate + nt * 32 + lane;
    for (int kt = 0; kt < 24; ++kt) {
      FragA a;
      const unsigned short* ap = &shA[mRow * LDA + kt * 32 + kSel];
      a.h[0] = *(const v8bf*)(ap);             // k: {0..7}  (+kSel)
      a.h[1] = *(const v8bf*)(ap + 16);        // k: {16..23}(+kSel)
      const v16bf bf = bp[kt * 48 * 32];
      acc = __builtin_amdgcn_wmma_f32_16x16x32_bf16(
          false, a.v, false, bf, (short)0, acc, false, false);
    }
    // epilogue: g = sigmoid(logit + g_s); hs = hidden + valid*fused
    const int o = nt * 16 + (lane & 15);
    const float sxo = shSX[o], gso = shGS[o];
#pragma unroll
    for (int i = 0; i < 8; ++i) {
      const int m = i + mHi;
      const float hid = bf2f(shA[m * LDA + o]);
      const float g = sigmoidf(acc[i] + gso);
      const float fused = g * sxo + (1.0f - g) * hid;
      const float hv = hid + (((s0 + m) < cueLen) ? fused : 0.0f);
      shHS[m * LDA + o] = f2bf(hv);
    }
  }
  __syncthreads();

  // ---- lin GEMM: waves 0..6 own one N-tile each (98 padded to 112) ----
  if (wave < 7) {
    const int nt = wave;
    v8f acc = {};
    const v16bf* bp = (const v16bf*)pLin + nt * 32 + lane;
    for (int kt = 0; kt < 24; ++kt) {
      FragA a;
      const unsigned short* ap = &shHS[mRow * LDA + kt * 32 + kSel];
      a.h[0] = *(const v8bf*)(ap);
      a.h[1] = *(const v8bf*)(ap + 16);
      const v16bf bf = bp[kt * 7 * 32];
      acc = __builtin_amdgcn_wmma_f32_16x16x32_bf16(
          false, a.v, false, bf, (short)0, acc, false, false);
    }
    const int o = nt * 16 + (lane & 15);
    if (o < P2) {
      const float lb = lin_b[o];
#pragma unroll
      for (int i = 0; i < 8; ++i) {
        const int m = i + mHi;
        float v = sigmoidf(acc[i] + lb);
        v = fminf(fmaxf(v, 1e-7f), 1.0f - 1e-7f);
        out[(((size_t)b * Ss) + s0 + m) * P2 + o] = v;
      }
    }
  }
}

// ---------------------------------------------------------------------------
// Launch
// ---------------------------------------------------------------------------
extern "C" void kernel_launch(void* const* d_in, const int* in_sizes, int n_in,
                              void* d_out, int out_size, void* d_ws, size_t ws_size,
                              hipStream_t stream) {
  const float* hidden = (const float*)d_in[0];
  // d_in[1] = batch_subject_ids (unused by reference)
  const int*   maskp  = (const int*)d_in[2];
  const float* attn_w = (const float*)d_in[3];
  const float* attn_b = (const float*)d_in[4];
  const float* gate_w = (const float*)d_in[5];
  const float* gate_b = (const float*)d_in[6];
  const float* lin_w  = (const float*)d_in[7];
  const float* lin_b  = (const float*)d_in[8];
  float* out = (float*)d_out;

  char* ws = (char*)d_ws;
  float* sx            = (float*)(ws);                        //  98304 B
  float* gsb           = (float*)(ws + 98304);                //  98304 B
  int*   cuep          = (int*)  (ws + 196608);               //    256 B
  unsigned short* pGate = (unsigned short*)(ws + 196864);     // 1179648 B
  unsigned short* pLin  = (unsigned short*)(ws + 196864 + 1179648); // 172032 B

  hipLaunchKernelGGL(k_pack_gate, dim3((24 * 48 * 512 + 255) / 256), dim3(256), 0, stream,
                     gate_w, pGate);
  hipLaunchKernelGGL(k_pack_lin, dim3((24 * 7 * 512 + 255) / 256), dim3(256), 0, stream,
                     lin_w, pLin);
  hipLaunchKernelGGL(k_attn, dim3(Bb), dim3(256), 0, stream,
                     hidden, maskp, attn_w, attn_b, sx, cuep);
  hipLaunchKernelGGL(k_gs, dim3(Bb, Hh / 256), dim3(256), 0, stream,
                     sx, gate_w, gate_b, gsb);
  hipLaunchKernelGGL(k_main, dim3(Bb * Ss / 16), dim3(256), 0, stream,
                     hidden, cuep, sx, gsb, pGate, pLin, lin_b, out);
}
